// MyGCN_35940286333262
// MI455X (gfx1250) — matrix operand order repaired and checked by reading, not compiled, not run on hardware
//
#include <hip/hip_runtime.h>

// ---------------------------------------------------------------------------
// GCN forward for MI455X (gfx1250, wave32, WMMA).
//   layer: h' = relu( h@W_in + scatter_add(norm * (h@W_neigh)[row] -> col) + b )
//   out  : h2 @ W_out + b_out
// Dense math: v_wmma_f32_16x16x32_bf16, dual-output strip GEMM (A read once,
// weights pre-packed to the per-lane B-operand layout -> b128 loads, no cvt).
// Edge phase: precomputed per-edge norm + float4 gather + f32 global atomics
// (destination buffer is L2-resident: 13 MB << 192 MB L2).
// ---------------------------------------------------------------------------

typedef __attribute__((ext_vector_type(16))) __bf16 v16bf;
typedef __attribute__((ext_vector_type(8)))  float  v8f;

static __device__ __forceinline__ unsigned short bf16_bits(float f) {
    union { float f; unsigned int u; } in; in.f = f;
    unsigned int u = in.u;
    u += 0x7fffu + ((u >> 16) & 1u);          // round-to-nearest-even
    return (unsigned short)(u >> 16);
}

union V16BF { v16bf v; unsigned short s[16]; };

// ---------------------------------------------------------------------------
// Pre-pack a [K x Hout] f32 weight matrix into bf16 in the exact per-lane
// v16bf B-operand layout:  Wp[((tn*KS + kk)*32 + lane)*16 + t]
//   lane: n = lane&15 (column), kbase = (lane>>4)*8
//   element t -> K row = kk*32 + kbase + (t&7) + (t>>3)*16,  column = tn*16 + n
// ---------------------------------------------------------------------------
__global__ __launch_bounds__(256) void pack_w_bf16(
    const float* __restrict__ W, unsigned short* __restrict__ Wp,
    int K, int Hout)
{
    int i = blockIdx.x * blockDim.x + threadIdx.x;
    if (i >= K * Hout) return;
    const int KS   = K >> 5;
    int t    = i & 15;
    int lane = (i >> 4) & 31;
    int tile = i >> 9;
    int kk   = tile % KS;
    int tn   = tile / KS;
    int krow = kk * 32 + ((lane >> 4) << 3) + (t & 7) + ((t >> 3) << 4);
    int col  = tn * 16 + (lane & 15);
    Wp[i] = bf16_bits(W[(size_t)krow * Hout + col]);
}

// ---------------------------------------------------------------------------
// Strip GEMM, dual weights sharing the A operand:
//   C1 = A @ W1, C2 = A @ W2      (A: n_rows x 32*KS, W: 32*KS x 16*TN)
// One wave32 per 16-row strip; A loaded once (float4), 2*TN WMMAs per k-step.
// ---------------------------------------------------------------------------
template<int TN, int KS>
__global__ __launch_bounds__(256) void gemm_dual_wmma(
    const float* __restrict__ A,
    const unsigned short* __restrict__ Wp1,
    const unsigned short* __restrict__ Wp2,
    float* __restrict__ C1, float* __restrict__ C2, int n_rows)
{
    const int K    = KS * 32;
    const int Hout = TN * 16;
    const int lane   = threadIdx.x & 31;
    const int waveId = blockIdx.x * (blockDim.x >> 5) + (threadIdx.x >> 5);
    if (waveId >= (n_rows >> 4)) return;       // whole-wave exit

    const int row0 = waveId << 4;
    const int m    = lane & 15;
    const int kh   = lane >> 4;

    v8f acc1[TN], acc2[TN];
#pragma unroll
    for (int j = 0; j < TN; ++j) { acc1[j] = (v8f){}; acc2[j] = (v8f){}; }

#pragma unroll
    for (int kk = 0; kk < KS; ++kk) {
        // A operand: 16x32 bf16 tile, contiguous rows -> 4x b128 loads
        const float* arow = A + (size_t)(row0 + m) * K + kk * 32 + kh * 8;
        float af[16];
        *(float4*)(af +  0) = *(const float4*)(arow +  0);   // K +0..3
        *(float4*)(af +  4) = *(const float4*)(arow +  4);   // K +4..7
        *(float4*)(af +  8) = *(const float4*)(arow + 16);   // K +16..19
        *(float4*)(af + 12) = *(const float4*)(arow + 20);   // K +20..23
        V16BF a;
#pragma unroll
        for (int t = 0; t < 16; ++t) a.s[t] = bf16_bits(af[t]);

#pragma unroll
        for (int j = 0; j < TN; ++j) {
            v16bf b = *(const v16bf*)(Wp1 + ((size_t)(j * KS + kk) * 32 + lane) * 16);
            acc1[j] = __builtin_amdgcn_wmma_f32_16x16x32_bf16(
                false, a.v, false, b, (short)0, acc1[j], false, false);
        }
#pragma unroll
        for (int j = 0; j < TN; ++j) {
            v16bf b = *(const v16bf*)(Wp2 + ((size_t)(j * KS + kk) * 32 + lane) * 16);
            acc2[j] = __builtin_amdgcn_wmma_f32_16x16x32_bf16(
                false, a.v, false, b, (short)0, acc2[j], false, false);
        }
    }

    // C/D layout: VGPR i -> M = i + 8*(lane>>4), N = lane&15
    const int orow = row0 + kh * 8;
#pragma unroll
    for (int j = 0; j < TN; ++j) {
        const int ocol = j * 16 + m;
#pragma unroll
        for (int i = 0; i < 8; ++i) {
            C1[(size_t)(orow + i) * Hout + ocol] = acc1[j][i];
            C2[(size_t)(orow + i) * Hout + ocol] = acc2[j][i];
        }
    }
}

// Single-weight strip GEMM with bias (output projection)
template<int TN, int KS>
__global__ __launch_bounds__(256) void gemm_single_wmma(
    const float* __restrict__ A, const unsigned short* __restrict__ Wp,
    const float* __restrict__ bias, float* __restrict__ C, int n_rows)
{
    const int K    = KS * 32;
    const int Hout = TN * 16;
    const int lane   = threadIdx.x & 31;
    const int waveId = blockIdx.x * (blockDim.x >> 5) + (threadIdx.x >> 5);
    if (waveId >= (n_rows >> 4)) return;

    const int row0 = waveId << 4;
    const int m    = lane & 15;
    const int kh   = lane >> 4;

    v8f acc[TN];
#pragma unroll
    for (int j = 0; j < TN; ++j) acc[j] = (v8f){};

#pragma unroll
    for (int kk = 0; kk < KS; ++kk) {
        const float* arow = A + (size_t)(row0 + m) * K + kk * 32 + kh * 8;
        float af[16];
        *(float4*)(af +  0) = *(const float4*)(arow +  0);
        *(float4*)(af +  4) = *(const float4*)(arow +  4);
        *(float4*)(af +  8) = *(const float4*)(arow + 16);
        *(float4*)(af + 12) = *(const float4*)(arow + 20);
        V16BF a;
#pragma unroll
        for (int t = 0; t < 16; ++t) a.s[t] = bf16_bits(af[t]);

#pragma unroll
        for (int j = 0; j < TN; ++j) {
            v16bf b = *(const v16bf*)(Wp + ((size_t)(j * KS + kk) * 32 + lane) * 16);
            acc[j] = __builtin_amdgcn_wmma_f32_16x16x32_bf16(
                false, a.v, false, b, (short)0, acc[j], false, false);
        }
    }

    const int orow = row0 + kh * 8;
#pragma unroll
    for (int j = 0; j < TN; ++j) {
        const int ocol = j * 16 + m;
        const float badd = bias[ocol];
#pragma unroll
        for (int i = 0; i < 8; ++i)
            C[(size_t)(orow + i) * Hout + ocol] = acc[j][i] + badd;
    }
}

// ---------------------------------------------------------------------------
// Degree / normalization helpers
// ---------------------------------------------------------------------------
__global__ __launch_bounds__(256) void zero_f32(float* __restrict__ p, int n) {
    int i = blockIdx.x * blockDim.x + threadIdx.x;
    if (i < n) p[i] = 0.0f;
}

__global__ __launch_bounds__(256) void deg_accum(const int* __restrict__ row,
                                                 float* __restrict__ deg, int E) {
    int i = blockIdx.x * blockDim.x + threadIdx.x;
    if (i < E) atomicAdd(&deg[row[i]], 1.0f);
}

__global__ __launch_bounds__(256) void deg_to_dinv(float* __restrict__ d, int n) {
    int i = blockIdx.x * blockDim.x + threadIdx.x;
    if (i < n) {
        float v = d[i];
        d[i] = (v > 0.0f) ? rsqrtf(v) : 0.0f;
    }
}

__global__ __launch_bounds__(256) void compute_norm(
    const int* __restrict__ row, const int* __restrict__ col,
    const float* __restrict__ dinv, float* __restrict__ norm, int E)
{
    int i = blockIdx.x * blockDim.x + threadIdx.x;
    if (i < E) norm[i] = dinv[row[i]] * dinv[col[i]];
}

// ---------------------------------------------------------------------------
// Edge message + scatter-add:  acc[col] += norm[e] * hn[row]
// 16 threads per edge, float4 per thread (H=64). Destination is L2-resident.
// ---------------------------------------------------------------------------
__global__ __launch_bounds__(256) void edge_scatter(
    const int* __restrict__ row, const int* __restrict__ col,
    const float* __restrict__ norm, const float* __restrict__ hn,
    float* __restrict__ acc, int E)
{
    int t = blockIdx.x * blockDim.x + threadIdx.x;
    int e = t >> 4;
    if (e >= E) return;
    int f = (t & 15) << 2;
    int r = row[e];
    int c = col[e];
    float nrm = norm[e];
    const float4 v = *(const float4*)(hn + (size_t)r * 64 + f);
    float* dst = acc + (size_t)c * 64 + f;
    atomicAdd(dst + 0, v.x * nrm);
    atomicAdd(dst + 1, v.y * nrm);
    atomicAdd(dst + 2, v.z * nrm);
    atomicAdd(dst + 3, v.w * nrm);
}

// h_out = relu(acc + bias)   (H == 64 -> feature index = i & 63)
__global__ __launch_bounds__(256) void bias_relu(
    const float* __restrict__ acc, const float* __restrict__ bias,
    float* __restrict__ out, int total)
{
    int i = blockIdx.x * blockDim.x + threadIdx.x;
    if (i < total) out[i] = fmaxf(acc[i] + bias[i & 63], 0.0f);
}

// ---------------------------------------------------------------------------
// Host-side launcher
// ---------------------------------------------------------------------------
extern "C" void kernel_launch(void* const* d_in, const int* in_sizes, int n_in,
                              void* d_out, int out_size, void* d_ws, size_t ws_size,
                              hipStream_t stream) {
    (void)n_in; (void)ws_size; (void)out_size;

    const float* x     = (const float*)d_in[0];
    const int*   eidx  = (const int*)  d_in[1];
    const float* W_in1 = (const float*)d_in[2];
    const float* W_ne1 = (const float*)d_in[3];
    const float* bias1 = (const float*)d_in[4];
    const float* W_in2 = (const float*)d_in[5];
    const float* W_ne2 = (const float*)d_in[6];
    const float* bias2 = (const float*)d_in[7];
    const float* W_out = (const float*)d_in[8];
    const float* b_out = (const float*)d_in[9];
    float*       out   = (float*)d_out;

    const int D = 64, H = 64, O = 32;
    const int N = in_sizes[0] / D;        // 50000
    const int E = in_sizes[1] / 2;        // 800000

    const int* erow = eidx;
    const int* ecol = eidx + E;

    // Workspace: dinv[N] | norm[E] | hn[N*H] | hi[N*H] | h[N*H] | packed weights
    float* dinv = (float*)d_ws;
    float* norm = dinv + N;
    float* hn   = norm + E;
    float* hi   = hn + (size_t)N * H;
    float* h    = hi + (size_t)N * H;
    uintptr_t wpa = ((uintptr_t)(h + (size_t)N * H) + 63) & ~(uintptr_t)63;
    unsigned short* wp_ne1 = (unsigned short*)wpa;
    unsigned short* wp_in1 = wp_ne1 + D * H;
    unsigned short* wp_ne2 = wp_in1 + D * H;
    unsigned short* wp_in2 = wp_ne2 + H * H;
    unsigned short* wp_out = wp_in2 + H * H;

    const int TB = 256;
    auto cdiv = [](long long a, long long b) { return (int)((a + b - 1) / b); };

    const int strip_blocks   = cdiv(N / 16, 8);          // 8 waves / block
    const int scatter_blocks = cdiv((long long)E * 16, TB);
    const int nh_blocks      = cdiv((long long)N * H, TB);

    // --- degree -> d^{-1/2} -> per-edge norm ---
    zero_f32    <<<cdiv(N, TB), TB, 0, stream>>>(dinv, N);
    deg_accum   <<<cdiv(E, TB), TB, 0, stream>>>(erow, dinv, E);
    deg_to_dinv <<<cdiv(N, TB), TB, 0, stream>>>(dinv, N);
    compute_norm<<<cdiv(E, TB), TB, 0, stream>>>(erow, ecol, dinv, norm, E);

    // --- pack weights to bf16 B-operand layout (cheap, every call) ---
    pack_w_bf16<<<cdiv(D * H, TB), TB, 0, stream>>>(W_ne1, wp_ne1, D, H);
    pack_w_bf16<<<cdiv(D * H, TB), TB, 0, stream>>>(W_in1, wp_in1, D, H);
    pack_w_bf16<<<cdiv(H * H, TB), TB, 0, stream>>>(W_ne2, wp_ne2, H, H);
    pack_w_bf16<<<cdiv(H * H, TB), TB, 0, stream>>>(W_in2, wp_in2, H, H);
    pack_w_bf16<<<cdiv(H * O, TB), TB, 0, stream>>>(W_out, wp_out, H, O);

    // --- layer 1 ---
    gemm_dual_wmma<4, 2><<<strip_blocks, TB, 0, stream>>>(x, wp_ne1, wp_in1, hn, hi, N);
    edge_scatter<<<scatter_blocks, TB, 0, stream>>>(erow, ecol, norm, hn, hi, E);
    bias_relu   <<<nh_blocks, TB, 0, stream>>>(hi, bias1, h, N * H);

    // --- layer 2 ---
    gemm_dual_wmma<4, 2><<<strip_blocks, TB, 0, stream>>>(h, wp_ne2, wp_in2, hn, hi, N);
    edge_scatter<<<scatter_blocks, TB, 0, stream>>>(erow, ecol, norm, hn, hi, E);
    bias_relu   <<<nh_blocks, TB, 0, stream>>>(hi, bias2, h, N * H);

    // --- output projection ---
    gemm_single_wmma<2, 2><<<strip_blocks, TB, 0, stream>>>(h, wp_out, b_out, out, N);
}
